// GraphSelfAttention_29154238005996
// MI455X (gfx1250) — matrix operand order repaired and checked by reading, not compiled, not run on hardware
//
#include <hip/hip_runtime.h>
#include <hip/hip_bf16.h>
#include <math.h>

#define DIMD  1024
#define HEADS 16
#define HDIM  64
#define BB    2
#define NSEQ  2048
#define BN    (BB * NSEQ)   // 4096

typedef __attribute__((ext_vector_type(16))) _Float16 v16h;
typedef __attribute__((ext_vector_type(8)))  _Float16 v8h;
typedef __attribute__((ext_vector_type(8)))  float    v8f;

__device__ __forceinline__ v8f wmma16(v16h a, v16h b, v8f c) {
  // D = A(16x32 f16) * B(32x16 f16) + C(16x16 f32)
  return __builtin_amdgcn_wmma_f32_16x16x32_f16(
      /*neg_a=*/false, a, /*neg_b=*/false, b,
      /*c_mod=*/(short)0, c, /*reuse_a=*/false, /*reuse_b=*/false);
}

// A fragment (16x32, f16) from row-major src. base points at [row0][k0], ld in halves.
// Lane layout (ISA 7.12.2): lanes 0-15: row=lane, K {0..7, 16..23}; lanes 16-31: row=lane-16, K {8..15, 24..31}.
__device__ __forceinline__ v16h load_matA(const _Float16* base, int ld, int lane) {
  const int l = lane & 15, hi = lane >> 4;
  const _Float16* p = base + (size_t)l * ld + hi * 8;
  v8h lo = *(const v8h*)(p);
  v8h hf = *(const v8h*)(p + 16);
  v16h a;
#pragma unroll
  for (int i = 0; i < 8; ++i) { a[i] = lo[i]; a[i + 8] = hf[i]; }
  return a;
}

// B fragment (32x16, f16): lanes 0-15: col=lane, K {0..7, 16..23}; lanes 16-31: col=lane-16, K {8..15, 24..31}.
// Source is row-major with "output column" as the row index: base points at [col0][k0].
__device__ __forceinline__ v16h load_matB(const _Float16* base, int ld, int lane) {
  const int l = lane & 15, hi = lane >> 4;
  const _Float16* p = base + (size_t)l * ld + hi * 8;
  v8h lo = *(const v8h*)(p);
  v8h hf = *(const v8h*)(p + 16);
  v16h b;
#pragma unroll
  for (int i = 0; i < 8; ++i) { b[i] = lo[i]; b[i + 8] = hf[i]; }
  return b;
}

// ---------------- fp32 -> f16 conversion ----------------
__global__ void cvt_f32_to_f16(const float* __restrict__ src, _Float16* __restrict__ dst, int n) {
  int i = blockIdx.x * blockDim.x + threadIdx.x;
  if (i < n) dst[i] = (_Float16)src[i];
}

// ---------------- QKV projection + fused per-head LayerNorm ----------------
// Each wave: 16 rows x 64 cols (one head). mode: 0/1 = LN + store [B,H,N,64]; 2 = no LN, store V transposed [B,H,64,N].
__global__ void __launch_bounds__(256)
proj_kernel(const _Float16* __restrict__ A,      // [BN][DIMD] f16
            const _Float16* __restrict__ W,      // [DIMD][DIMD] f16 row-major (e, d)
            const float*    __restrict__ bias,   // [DIMD]
            const float*    __restrict__ gamma,  // [HDIM]
            const float*    __restrict__ beta,   // [HDIM]
            _Float16*       __restrict__ dst,
            int mode)
{
  const int lane = threadIdx.x & 31;
  const int wave = threadIdx.x >> 5;
  const int l = lane & 15, hi = lane >> 4;
  const int m0 = blockIdx.x * 128 + wave * 16;   // row base in [0, BN)
  const int h  = blockIdx.y;                     // head
  const int e0 = h * HDIM;                       // column base

  v8f c0 = {}, c1 = {}, c2 = {}, c3 = {};
  for (int k0 = 0; k0 < DIMD; k0 += 32) {
    v16h a  = load_matA(A + (size_t)m0 * DIMD + k0, DIMD, lane);
    v16h b0 = load_matB(W + (size_t)(e0     ) * DIMD + k0, DIMD, lane);
    v16h b1 = load_matB(W + (size_t)(e0 + 16) * DIMD + k0, DIMD, lane);
    v16h b2 = load_matB(W + (size_t)(e0 + 32) * DIMD + k0, DIMD, lane);
    v16h b3 = load_matB(W + (size_t)(e0 + 48) * DIMD + k0, DIMD, lane);
    c0 = wmma16(a, b0, c0);
    c1 = wmma16(a, b1, c1);
    c2 = wmma16(a, b2, c2);
    c3 = wmma16(a, b3, c3);
  }

  const float bj0 = bias[e0 + l], bj1 = bias[e0 + 16 + l];
  const float bj2 = bias[e0 + 32 + l], bj3 = bias[e0 + 48 + l];
#pragma unroll
  for (int r = 0; r < 8; ++r) { c0[r] += bj0; c1[r] += bj1; c2[r] += bj2; c3[r] += bj3; }

  if (mode < 2) {
    const float g0 = gamma[l], g1 = gamma[16 + l], g2 = gamma[32 + l], g3 = gamma[48 + l];
    const float t0 = beta[l],  t1 = beta[16 + l],  t2 = beta[32 + l],  t3 = beta[48 + l];
#pragma unroll
    for (int r = 0; r < 8; ++r) {
      float s = c0[r] + c1[r] + c2[r] + c3[r];
      float q = c0[r]*c0[r] + c1[r]*c1[r] + c2[r]*c2[r] + c3[r]*c3[r];
#pragma unroll
      for (int m = 1; m < 16; m <<= 1) {
        s += __shfl_xor(s, m, 32);
        q += __shfl_xor(q, m, 32);
      }
      const float mu  = s * (1.0f / 64.0f);
      const float var = q * (1.0f / 64.0f) - mu * mu;
      const float rs  = rsqrtf(var + 1e-6f);
      c0[r] = (c0[r] - mu) * rs * g0 + t0;
      c1[r] = (c1[r] - mu) * rs * g1 + t1;
      c2[r] = (c2[r] - mu) * rs * g2 + t2;
      c3[r] = (c3[r] - mu) * rs * g3 + t3;
    }
  }

#pragma unroll
  for (int r = 0; r < 8; ++r) {
    const int n  = m0 + r + 8 * hi;
    const int b  = n >> 11;            // / NSEQ
    const int nn = n & (NSEQ - 1);
    if (mode < 2) {
      _Float16* p = dst + ((size_t)(b * HEADS + h) * NSEQ + nn) * HDIM;
      p[l]      = (_Float16)c0[r];
      p[16 + l] = (_Float16)c1[r];
      p[32 + l] = (_Float16)c2[r];
      p[48 + l] = (_Float16)c3[r];
    } else {  // V transposed: [B,H,64,N]
      _Float16* p = dst + (size_t)(b * HEADS + h) * HDIM * NSEQ + nn;
      p[(size_t)(l)      * NSEQ] = (_Float16)c0[r];
      p[(size_t)(16 + l) * NSEQ] = (_Float16)c1[r];
      p[(size_t)(32 + l) * NSEQ] = (_Float16)c2[r];
      p[(size_t)(48 + l) * NSEQ] = (_Float16)c3[r];
    }
  }
}

// ---------------- flash attention: one wave per 16-query tile ----------------
__global__ void __launch_bounds__(256)
attn_kernel(const _Float16* __restrict__ Qh,   // [B*H][N][64]
            const _Float16* __restrict__ Kh,   // [B*H][N][64]
            const _Float16* __restrict__ Vt,   // [B*H][64][N]
            const float*    __restrict__ bias, // [B][N][N]
            _Float16*       __restrict__ AO)   // [B][N][DIMD]
{
  __shared__ __align__(64) _Float16 Pl[8][16][32];  // per-wave P transpose staging

  const int lane = threadIdx.x & 31;
  const int wave = threadIdx.x >> 5;
  const int l = lane & 15, hi = lane >> 4;
  const int tile = blockIdx.x * 8 + wave;
  const int bh = tile >> 7;                 // / (N/16)
  const int q0 = (tile & 127) * 16;
  const int b  = bh >> 4, h = bh & 15;

  const _Float16* Qb = Qh + (size_t)bh * NSEQ * HDIM;
  const _Float16* Kb = Kh + (size_t)bh * NSEQ * HDIM;
  const _Float16* Vb = Vt + (size_t)bh * HDIM * NSEQ;
  const float*    Bb = bias + (size_t)b * NSEQ * NSEQ;

  const v16h aq0 = load_matA(Qb + (size_t)q0 * HDIM + 0,  HDIM, lane);
  const v16h aq1 = load_matA(Qb + (size_t)q0 * HDIM + 32, HDIM, lane);

  v8f o0 = {}, o1 = {}, o2 = {}, o3 = {};
  float rmax[8], rsum[8];
#pragma unroll
  for (int r = 0; r < 8; ++r) { rmax[r] = -INFINITY; rsum[r] = 0.0f; }

  for (int m0 = 0; m0 < NSEQ; m0 += 32) {
    // S = Q * K^T for 32 keys (K rows are directly column-major B fragments)
    v16h bk00 = load_matB(Kb + (size_t)(m0     ) * HDIM + 0,  HDIM, lane);
    v16h bk01 = load_matB(Kb + (size_t)(m0     ) * HDIM + 32, HDIM, lane);
    v16h bk10 = load_matB(Kb + (size_t)(m0 + 16) * HDIM + 0,  HDIM, lane);
    v16h bk11 = load_matB(Kb + (size_t)(m0 + 16) * HDIM + 32, HDIM, lane);
    v8f s0 = {}, s1 = {};
    s0 = wmma16(aq0, bk00, s0); s0 = wmma16(aq1, bk01, s0);
    s1 = wmma16(aq0, bk10, s1); s1 = wmma16(aq1, bk11, s1);

    if (m0 + 32 < NSEQ)  // stream attn_bias through L2: prefetch next tile
      __builtin_prefetch(Bb + (size_t)(q0 + 8 * hi) * NSEQ + m0 + 32, 0, 0);

    float p0[8], p1[8];
#pragma unroll
    for (int r = 0; r < 8; ++r) {
      const float* brow = Bb + (size_t)(q0 + r + 8 * hi) * NSEQ + m0;
      p0[r] = s0[r] * 0.125f + brow[l];
      p1[r] = s1[r] * 0.125f + brow[16 + l];
    }

#pragma unroll
    for (int r = 0; r < 8; ++r) {
      float tm = fmaxf(p0[r], p1[r]);
#pragma unroll
      for (int m = 1; m < 16; m <<= 1) tm = fmaxf(tm, __shfl_xor(tm, m, 32));
      const float nm   = fmaxf(rmax[r], tm);
      const float corr = __expf(rmax[r] - nm);
      rmax[r] = nm;
      p0[r] = __expf(p0[r] - nm);
      p1[r] = __expf(p1[r] - nm);
      float ts = p0[r] + p1[r];
#pragma unroll
      for (int m = 1; m < 16; m <<= 1) ts += __shfl_xor(ts, m, 32);
      rsum[r] = rsum[r] * corr + ts;
      o0[r] *= corr; o1[r] *= corr; o2[r] *= corr; o3[r] *= corr;
      // stage P row-major in LDS (C layout -> A layout transpose); per-wave, DS ops are in-order
      Pl[wave][r + 8 * hi][l]      = (_Float16)p0[r];
      Pl[wave][r + 8 * hi][16 + l] = (_Float16)p1[r];
    }

    const v16h pa  = load_matA(&Pl[wave][0][0], 32, lane);
    v16h bv0 = load_matB(Vb + (size_t)( 0) * NSEQ + m0, NSEQ, lane);
    v16h bv1 = load_matB(Vb + (size_t)(16) * NSEQ + m0, NSEQ, lane);
    v16h bv2 = load_matB(Vb + (size_t)(32) * NSEQ + m0, NSEQ, lane);
    v16h bv3 = load_matB(Vb + (size_t)(48) * NSEQ + m0, NSEQ, lane);
    o0 = wmma16(pa, bv0, o0);
    o1 = wmma16(pa, bv1, o1);
    o2 = wmma16(pa, bv2, o2);
    o3 = wmma16(pa, bv3, o3);
  }

#pragma unroll
  for (int r = 0; r < 8; ++r) {
    const float inv = 1.0f / rsum[r];
    const int n = q0 + r + 8 * hi;
    _Float16* p = AO + ((size_t)b * NSEQ + n) * DIMD + h * HDIM;
    p[l]      = (_Float16)(o0[r] * inv);
    p[16 + l] = (_Float16)(o1[r] * inv);
    p[32 + l] = (_Float16)(o2[r] * inv);
    p[48 + l] = (_Float16)(o3[r] * inv);
  }
}

// ---------------- output projection (fp32 out) ----------------
__global__ void __launch_bounds__(256)
outproj_kernel(const _Float16* __restrict__ A,   // [BN][DIMD] f16
               const _Float16* __restrict__ W,   // [DIMD][DIMD] f16 row-major
               const float*    __restrict__ bias,
               float*          __restrict__ out) // [BN][DIMD] f32
{
  const int lane = threadIdx.x & 31;
  const int wave = threadIdx.x >> 5;
  const int l = lane & 15, hi = lane >> 4;
  const int m0 = blockIdx.x * 128 + wave * 16;
  const int e0 = blockIdx.y * 64;

  v8f c0 = {}, c1 = {}, c2 = {}, c3 = {};
  for (int k0 = 0; k0 < DIMD; k0 += 32) {
    v16h a = load_matA(A + (size_t)m0 * DIMD + k0, DIMD, lane);
    c0 = wmma16(a, load_matB(W + (size_t)(e0     ) * DIMD + k0, DIMD, lane), c0);
    c1 = wmma16(a, load_matB(W + (size_t)(e0 + 16) * DIMD + k0, DIMD, lane), c1);
    c2 = wmma16(a, load_matB(W + (size_t)(e0 + 32) * DIMD + k0, DIMD, lane), c2);
    c3 = wmma16(a, load_matB(W + (size_t)(e0 + 48) * DIMD + k0, DIMD, lane), c3);
  }
  const float b0 = bias[e0 + l], b1 = bias[e0 + 16 + l];
  const float b2 = bias[e0 + 32 + l], b3 = bias[e0 + 48 + l];
#pragma unroll
  for (int r = 0; r < 8; ++r) {
    float* p = out + (size_t)(m0 + r + 8 * hi) * DIMD + e0;
    p[l]      = c0[r] + b0;
    p[16 + l] = c1[r] + b1;
    p[32 + l] = c2[r] + b2;
    p[48 + l] = c3[r] + b3;
  }
}

extern "C" void kernel_launch(void* const* d_in, const int* in_sizes, int n_in,
                              void* d_out, int out_size, void* d_ws, size_t ws_size,
                              hipStream_t stream) {
  const float* x    = (const float*)d_in[0];
  const float* bias = (const float*)d_in[1];
  const float* Wq   = (const float*)d_in[2];
  const float* bq   = (const float*)d_in[3];
  const float* Wk   = (const float*)d_in[4];
  const float* bk   = (const float*)d_in[5];
  const float* Wv   = (const float*)d_in[6];
  const float* bv   = (const float*)d_in[7];
  const float* Wo   = (const float*)d_in[8];
  const float* bo   = (const float*)d_in[9];
  const float* qg   = (const float*)d_in[10];
  const float* qb   = (const float*)d_in[11];
  const float* kg   = (const float*)d_in[12];
  const float* kb   = (const float*)d_in[13];
  float* out = (float*)d_out;

  char* ws = (char*)d_ws;
  const size_t MB = 1024ull * 1024ull;
  _Float16* xh  = (_Float16*)(ws);             //  8 MB: [BN][D]
  _Float16* wqh = (_Float16*)(ws +  8 * MB);   //  2 MB
  _Float16* wkh = (_Float16*)(ws + 10 * MB);   //  2 MB
  _Float16* wvh = (_Float16*)(ws + 12 * MB);   //  2 MB
  _Float16* woh = (_Float16*)(ws + 14 * MB);   //  2 MB
  _Float16* Qh  = (_Float16*)(ws + 16 * MB);   //  8 MB: [B*H][N][64]
  _Float16* Kh  = (_Float16*)(ws + 24 * MB);   //  8 MB
  _Float16* Vt  = (_Float16*)(ws + 32 * MB);   //  8 MB: [B*H][64][N]
  _Float16* AO  = (_Float16*)(ws + 40 * MB);   //  8 MB: [B][N][D]

  const int nx = BN * DIMD, nw = DIMD * DIMD;
  cvt_f32_to_f16<<<(nx + 255) / 256, 256, 0, stream>>>(x,  xh,  nx);
  cvt_f32_to_f16<<<(nw + 255) / 256, 256, 0, stream>>>(Wq, wqh, nw);
  cvt_f32_to_f16<<<(nw + 255) / 256, 256, 0, stream>>>(Wk, wkh, nw);
  cvt_f32_to_f16<<<(nw + 255) / 256, 256, 0, stream>>>(Wv, wvh, nw);
  cvt_f32_to_f16<<<(nw + 255) / 256, 256, 0, stream>>>(Wo, woh, nw);

  dim3 gProj(BN / 128, HEADS);
  proj_kernel<<<gProj, 256, 0, stream>>>(xh, wqh, bq, qg, qb, Qh, 0);
  proj_kernel<<<gProj, 256, 0, stream>>>(xh, wkh, bk, kg, kb, Kh, 1);
  proj_kernel<<<gProj, 256, 0, stream>>>(xh, wvh, bv, qg, qb, Vt, 2);

  attn_kernel<<<(BB * HEADS * (NSEQ / 16)) / 8, 256, 0, stream>>>(Qh, Kh, Vt, bias, AO);

  outproj_kernel<<<dim3(BN / 128, DIMD / 64), 256, 0, stream>>>(AO, woh, bo, out);
}
